// HGCNIISolver_11081015623839
// MI455X (gfx1250) — matrix operand order repaired and checked by reading, not compiled, not run on hardware
//
#include <hip/hip_runtime.h>
#include <math.h>

#define NVN 100000
#define NCN 420000
#define NEN 630000
#define HD  128
#define LN  8
#define HH  (HD*HD)
#define ALPHA_C 0.1f
#define THETA_C 0.5f

typedef __attribute__((ext_vector_type(16))) __bf16 v16bf;
typedef __attribute__((ext_vector_type(8)))  __bf16 v8bf;
typedef __attribute__((ext_vector_type(8)))  float  v8f;

static __device__ __forceinline__ unsigned short f2bf(float f) {
  unsigned int u = __float_as_uint(f);
  u += 0x7FFFu + ((u >> 16) & 1u);
  return (unsigned short)(u >> 16);
}

// ---------------- embeddings ----------------
__global__ void k_emb_var(const float* __restrict__ x, const float* __restrict__ W,
                          const float* __restrict__ b, float* __restrict__ h0,
                          float* __restrict__ h) {
  int idx = blockIdx.x * blockDim.x + threadIdx.x;
  if (idx >= NVN * HD) return;
  int n = idx >> 7, c = idx & 127;
  const float* xp = x + n * 4;
  float acc = b[c];
#pragma unroll
  for (int k = 0; k < 4; ++k) acc += xp[k] * W[k * HD + c];
  acc = fmaxf(acc, 0.f);
  h0[idx] = acc; h[idx] = acc;
}

__global__ void k_emb_cla(const float* __restrict__ x, const float* __restrict__ W,
                          const float* __restrict__ b, float* __restrict__ h0,
                          float* __restrict__ h) {
  int idx = blockIdx.x * blockDim.x + threadIdx.x;
  if (idx >= NCN * HD) return;
  int n = idx >> 7, c = idx & 127;
  float acc = fmaxf(x[n] * W[c] + b[c], 0.f);
  h0[idx] = acc; h[idx] = acc;
}

// ---------------- degree counts / inverse ----------------
__global__ void k_count(const int* __restrict__ idx, float* __restrict__ cnt, int nE) {
  int e = blockIdx.x * blockDim.x + threadIdx.x;
  if (e < nE)
    __hip_atomic_fetch_add(&cnt[idx[e]], 1.0f, __ATOMIC_RELAXED, __HIP_MEMORY_SCOPE_AGENT);
}

__global__ void k_inv(float* __restrict__ c, int n) {
  int i = blockIdx.x * blockDim.x + threadIdx.x;
  if (i < n) c[i] = 1.0f / fmaxf(c[i], 1.0f);
}

// ---------------- edge scatter-add (segment sum), one wave per edge ----------------
__global__ void k_scatter(const float* __restrict__ feat, const int* __restrict__ s,
                          const int* __restrict__ d, float* __restrict__ agg, int nE) {
  int e = blockIdx.x * 8 + (threadIdx.x >> 5);
  if (e >= nE) return;
  int lane = threadIdx.x & 31;
  const float4 v = *(const float4*)(feat + (size_t)s[e] * HD + lane * 4);
  float* out = agg + (size_t)d[e] * HD + lane * 4;
  __hip_atomic_fetch_add(out + 0, v.x, __ATOMIC_RELAXED, __HIP_MEMORY_SCOPE_AGENT);
  __hip_atomic_fetch_add(out + 1, v.y, __ATOMIC_RELAXED, __HIP_MEMORY_SCOPE_AGENT);
  __hip_atomic_fetch_add(out + 2, v.z, __ATOMIC_RELAXED, __HIP_MEMORY_SCOPE_AGENT);
  __hip_atomic_fetch_add(out + 3, v.w, __ATOMIC_RELAXED, __HIP_MEMORY_SCOPE_AGENT);
}

// ---------------- f32 -> bf16 (optional per-row mean scaling) ----------------
__global__ void k_to_bf16(const float* __restrict__ x, const float* __restrict__ rowscale,
                          unsigned short* __restrict__ y, int n) {
  int i = blockIdx.x * blockDim.x + threadIdx.x;
  if (i >= n) return;
  float s = rowscale ? rowscale[i >> 7] : 1.0f;
  y[i] = f2bf(x[i] * s);
}

// ---- WMMA B-fragment packing: element W[r][c] (r = K index, c = N column) goes to
//      packed[((kcIdx*2 + kg)*128 + c)*16 + ii], kcIdx=r>>5, kg=(r>>4)&1, ii=r&15.
//      So each lane's 16 K-values for its column are one contiguous 32B span. ----
static __device__ __forceinline__ size_t bpack_idx(int r, int c) {
  int kcIdx = r >> 5, kg = (r >> 4) & 1, ii = r & 15;
  return (((size_t)(kcIdx * 2 + kg) * HD) + c) * 16 + ii;
}

// per-layer weight concat [Wl_a ; Wl_b ; Wr_a+Wr_b] -> bf16, pre-packed
__global__ void k_wcat(const float* __restrict__ Wl_a, const float* __restrict__ Wl_b,
                       const float* __restrict__ Wr_a, const float* __restrict__ Wr_b,
                       const float* __restrict__ bl_a, const float* __restrict__ bl_b,
                       unsigned short* __restrict__ Wout, float* __restrict__ bout) {
  int i = blockIdx.x * blockDim.x + threadIdx.x;
  if (i >= 3 * HH) return;
  int r = i >> 7, c = i & 127;
  float w;
  if (r < HD)            w = Wl_a[r * HD + c];
  else if (r < 2 * HD)   w = Wl_b[(r - HD) * HD + c];
  else                   w = Wr_a[(r - 2 * HD) * HD + c] + Wr_b[(r - 2 * HD) * HD + c];
  Wout[bpack_idx(r, c)] = f2bf(w);
  if (i < HD) bout[i] = bl_a[i] + bl_b[i];
}

// single HxH weight -> bf16, pre-packed
__global__ void k_wpack(const float* __restrict__ W, unsigned short* __restrict__ out) {
  int i = blockIdx.x * blockDim.x + threadIdx.x;
  if (i >= HH) return;
  int r = i >> 7, c = i & 127;
  out[bpack_idx(r, c)] = f2bf(W[i]);
}

// ---------------- fused bf16 WMMA GEMM (persistent, B register-resident) ----------------
// Block = 256 thr = 8 waves; wave w owns columns 16w..16w+15 and preloads its B fragments
// once. Block grid-strides over 16-row tiles of the output.
// t = w0*(acc+bias) + w1*aux0[idx]  (+ aux1[idx])  (+ relu); writes f32 and optional bf16.
template <int NSRC>
__global__ __launch_bounds__(256) void k_wmma_gemm(
    const __bf16* X0, const __bf16* X1, const __bf16* X2,
    const __bf16* __restrict__ W, const float* __restrict__ bias,
    const float* aux0, const float* aux1,
    float* outF, unsigned short* outB,
    int ntiles, float w0, float w1, int doRelu) {
  const int lane = threadIdx.x & 31;
  const int wave = threadIdx.x >> 5;
  const int m  = lane & 15;
  const int hi = lane >> 4;          // lane half-group
  const int kb = hi * 16;            // A fragment K sub-offset
  const int n  = wave * 16 + m;      // output column

  // Preload all B fragments for this wave's column tile (packed layout -> 32B vector loads)
  v16bf bfrag[NSRC * 4];
#pragma unroll
  for (int kcIdx = 0; kcIdx < NSRC * 4; ++kcIdx) {
    const __bf16* pb = W + (((size_t)(kcIdx * 2 + hi) * HD) + n) * 16;
    bfrag[kcIdx] = *(const v16bf*)pb;
  }
  const float bcol = bias[n];
  const int rbase = hi * 8;

  for (int t = blockIdx.x; t < ntiles; t += gridDim.x) {
    const int row0 = t * 16;
    if (t + (int)gridDim.x < ntiles)
      __builtin_prefetch(X0 + (size_t)(t + gridDim.x) * 16 * HD, 0, 1);
    v8f acc = {};
#pragma unroll
    for (int s = 0; s < NSRC; ++s) {
      const __bf16* Xp = (s == 0 ? X0 : (s == 1 ? X1 : X2)) + (size_t)(row0 + m) * HD;
#pragma unroll
      for (int kc4 = 0; kc4 < 4; ++kc4) {
        // A: 16x32 bf16, row m, lane-group selects K halves (contiguous k = kb + s, same order as B packing)
        v8bf lo = *(const v8bf*)(Xp + kc4 * 32 + kb);        // K = kb .. kb+7
        v8bf hg = *(const v8bf*)(Xp + kc4 * 32 + kb + 8);    // K = kb+8 .. kb+15
        v16bf a;
#pragma unroll
        for (int i = 0; i < 8; ++i) { a[i] = lo[i]; a[8 + i] = hg[i]; }
        acc = __builtin_amdgcn_wmma_f32_16x16x32_bf16(false, a, false, bfrag[s * 4 + kc4],
                                                      (short)0, acc, false, false);
      }
    }
    // Epilogue: C/D layout — acc[r]: row = 8*hi + r, col = n
#pragma unroll
    for (int r = 0; r < 8; ++r) {
      size_t idx = (size_t)(row0 + rbase + r) * HD + n;
      float tv = w0 * (acc[r] + bcol) + w1 * aux0[idx];
      if (aux1) tv += aux1[idx];
      if (doRelu) tv = fmaxf(tv, 0.f);
      outF[idx] = tv;
      if (outB) outB[idx] = f2bf(tv);
    }
  }
}

// ---------------- final projection: out = v_h @ W_fin + b_fin ----------------
__global__ void k_final(const float* __restrict__ vh, const float* __restrict__ Wf,
                        const float* __restrict__ bf, float* __restrict__ out) {
  int nidx = blockIdx.x * blockDim.x + threadIdx.x;
  if (nidx >= NVN) return;
  float acc = bf[0];
  const float* p = vh + (size_t)nidx * HD;
#pragma unroll 8
  for (int k = 0; k < HD; ++k) acc += p[k] * Wf[k];
  out[nidx] = acc;
}

extern "C" void kernel_launch(void* const* d_in, const int* in_sizes, int n_in,
                              void* d_out, int out_size, void* d_ws, size_t ws_size,
                              hipStream_t stream) {
  (void)in_sizes; (void)n_in; (void)out_size; (void)ws_size;
  const float* x_variable = (const float*)d_in[0];
  const float* x_clause   = (const float*)d_in[1];
  const int*   edge_pos   = (const int*)d_in[2];
  const int*   edge_neg   = (const int*)d_in[3];
  const float* W_vemb = (const float*)d_in[4];
  const float* b_vemb = (const float*)d_in[5];
  const float* W_cemb = (const float*)d_in[6];
  const float* b_cemb = (const float*)d_in[7];
  const float* Wl_pos = (const float*)d_in[8];
  const float* bl_pos = (const float*)d_in[9];
  const float* Wr_pos = (const float*)d_in[10];
  const float* Wl_neg = (const float*)d_in[11];
  const float* bl_neg = (const float*)d_in[12];
  const float* Wr_neg = (const float*)d_in[13];
  const float* Wl_rpos = (const float*)d_in[14];
  const float* bl_rpos = (const float*)d_in[15];
  const float* Wr_rpos = (const float*)d_in[16];
  const float* Wl_rneg = (const float*)d_in[17];
  const float* bl_rneg = (const float*)d_in[18];
  const float* Wr_rneg = (const float*)d_in[19];
  const float* Wv    = (const float*)d_in[20];
  const float* bv    = (const float*)d_in[21];
  const float* W_fin = (const float*)d_in[22];
  const float* b_fin = (const float*)d_in[23];

  const int* sp = edge_pos;        const int* dp = edge_pos + NEN;
  const int* sn = edge_neg;        const int* dn = edge_neg + NEN;

  // ---- workspace bump allocator (256B aligned) ----
  char* base = (char*)d_ws;
  size_t off = 0;
  auto alloc = [&](size_t bytes) -> void* {
    void* r = base + off;
    off = (off + bytes + 255) & ~(size_t)255;
    return r;
  };
  float* v_h0  = (float*)alloc((size_t)NVN * HD * 4);
  float* v_h   = (float*)alloc((size_t)NVN * HD * 4);
  float* v_tmp = (float*)alloc((size_t)NVN * HD * 4);
  float* c_h0  = (float*)alloc((size_t)NCN * HD * 4);
  float* c_h   = (float*)alloc((size_t)NCN * HD * 4);
  float* aggPc = (float*)alloc((size_t)NCN * HD * 4);
  float* aggNc = (float*)alloc((size_t)NCN * HD * 4);
  float* aggPv = (float*)alloc((size_t)NVN * HD * 4);
  float* aggNv = (float*)alloc((size_t)NVN * HD * 4);
  unsigned short* aggPc_b = (unsigned short*)alloc((size_t)NCN * HD * 2);
  unsigned short* aggNc_b = (unsigned short*)alloc((size_t)NCN * HD * 2);
  unsigned short* ch_b    = (unsigned short*)alloc((size_t)NCN * HD * 2);
  unsigned short* aggPv_b = (unsigned short*)alloc((size_t)NVN * HD * 2);
  unsigned short* aggNv_b = (unsigned short*)alloc((size_t)NVN * HD * 2);
  unsigned short* vh_b    = (unsigned short*)alloc((size_t)NVN * HD * 2);
  unsigned short* v1_b    = (unsigned short*)alloc((size_t)NVN * HD * 2);
  float* cntPc = (float*)alloc((size_t)NCN * 4);   // holds inverse counts after k_inv
  float* cntNc = (float*)alloc((size_t)NCN * 4);
  float* cntPv = (float*)alloc((size_t)NVN * 4);
  float* cntNv = (float*)alloc((size_t)NVN * 4);
  unsigned short* WcCat = (unsigned short*)alloc((size_t)LN * 3 * HH * 2);
  unsigned short* WvCat = (unsigned short*)alloc((size_t)LN * 3 * HH * 2);
  unsigned short* Wvm   = (unsigned short*)alloc((size_t)LN * HH * 2);
  float* biasC = (float*)alloc((size_t)LN * HD * 4);
  float* biasV = (float*)alloc((size_t)LN * HD * 4);

  const int TB = 256;
  const int gVH = (NVN * HD + TB - 1) / TB;
  const int gCH = (NCN * HD + TB - 1) / TB;
  const int gE  = (NEN + TB - 1) / TB;
  const int gEw = (NEN + 7) / 8;          // one wave per edge

  const int tilesC = NCN / 16, tilesV = NVN / 16;
  const int gGemmC = tilesC < 2048 ? tilesC : 2048;
  const int gGemmV = tilesV < 2048 ? tilesV : 2048;

  // ---- degree counts -> inverse means (layer invariant) ----
  hipMemsetAsync(cntPc, 0, (size_t)NCN * 4, stream);
  hipMemsetAsync(cntNc, 0, (size_t)NCN * 4, stream);
  hipMemsetAsync(cntPv, 0, (size_t)NVN * 4, stream);
  hipMemsetAsync(cntNv, 0, (size_t)NVN * 4, stream);
  k_count<<<gE, TB, 0, stream>>>(dp, cntPc, NEN);
  k_count<<<gE, TB, 0, stream>>>(dn, cntNc, NEN);
  k_count<<<gE, TB, 0, stream>>>(sp, cntPv, NEN);
  k_count<<<gE, TB, 0, stream>>>(sn, cntNv, NEN);
  k_inv<<<(NCN + TB - 1) / TB, TB, 0, stream>>>(cntPc, NCN);
  k_inv<<<(NCN + TB - 1) / TB, TB, 0, stream>>>(cntNc, NCN);
  k_inv<<<(NVN + TB - 1) / TB, TB, 0, stream>>>(cntPv, NVN);
  k_inv<<<(NVN + TB - 1) / TB, TB, 0, stream>>>(cntNv, NVN);

  // ---- per-layer weight prep (bf16 packed weights, combined biases) ----
  const int gW = (3 * HH + TB - 1) / TB;
  for (int i = 0; i < LN; ++i) {
    k_wcat<<<gW, TB, 0, stream>>>(Wl_pos + (size_t)i * HH, Wl_neg + (size_t)i * HH,
                                  Wr_pos + (size_t)i * HH, Wr_neg + (size_t)i * HH,
                                  bl_pos + (size_t)i * HD, bl_neg + (size_t)i * HD,
                                  WcCat + (size_t)i * 3 * HH, biasC + (size_t)i * HD);
    k_wcat<<<gW, TB, 0, stream>>>(Wl_rpos + (size_t)i * HH, Wl_rneg + (size_t)i * HH,
                                  Wr_rpos + (size_t)i * HH, Wr_rneg + (size_t)i * HH,
                                  bl_rpos + (size_t)i * HD, bl_rneg + (size_t)i * HD,
                                  WvCat + (size_t)i * 3 * HH, biasV + (size_t)i * HD);
    k_wpack<<<(HH + TB - 1) / TB, TB, 0, stream>>>(Wv + (size_t)i * HH,
                                                   Wvm + (size_t)i * HH);
  }

  // ---- embeddings ----
  k_emb_var<<<gVH, TB, 0, stream>>>(x_variable, W_vemb, b_vemb, v_h0, v_h);
  k_emb_cla<<<gCH, TB, 0, stream>>>(x_clause, W_cemb, b_cemb, c_h0, c_h);
  k_to_bf16<<<gCH, TB, 0, stream>>>(c_h, nullptr, ch_b, NCN * HD);
  k_to_bf16<<<gVH, TB, 0, stream>>>(v_h, nullptr, vh_b, NVN * HD);

  // ---- layers ----
  for (int i = 0; i < LN; ++i) {
    hipMemsetAsync(aggPc, 0, (size_t)NCN * HD * 4, stream);
    hipMemsetAsync(aggNc, 0, (size_t)NCN * HD * 4, stream);
    hipMemsetAsync(aggPv, 0, (size_t)NVN * HD * 4, stream);
    hipMemsetAsync(aggNv, 0, (size_t)NVN * HD * 4, stream);

    k_scatter<<<gEw, TB, 0, stream>>>(v_h, sp, dp, aggPc, NEN);  // var -> clause (pos)
    k_scatter<<<gEw, TB, 0, stream>>>(v_h, sn, dn, aggNc, NEN);  // var -> clause (neg)
    k_scatter<<<gEw, TB, 0, stream>>>(c_h, dp, sp, aggPv, NEN);  // clause -> var (pos)
    k_scatter<<<gEw, TB, 0, stream>>>(c_h, dn, sn, aggNv, NEN);  // clause -> var (neg)

    k_to_bf16<<<gCH, TB, 0, stream>>>(aggPc, cntPc, aggPc_b, NCN * HD);
    k_to_bf16<<<gCH, TB, 0, stream>>>(aggNc, cntNc, aggNc_b, NCN * HD);
    k_to_bf16<<<gVH, TB, 0, stream>>>(aggPv, cntPv, aggPv_b, NVN * HD);
    k_to_bf16<<<gVH, TB, 0, stream>>>(aggNv, cntNv, aggNv_b, NVN * HD);

    // clause: c_h = relu((1-a)*([aggP|aggN|c_h]@Wc + bC) + a*c_h0); emit f32 + bf16
    k_wmma_gemm<3><<<gGemmC, 256, 0, stream>>>(
        (const __bf16*)aggPc_b, (const __bf16*)aggNc_b, (const __bf16*)ch_b,
        (const __bf16*)(WcCat + (size_t)i * 3 * HH), biasC + (size_t)i * HD,
        c_h0, nullptr, c_h, ch_b, tilesC, 1.0f - ALPHA_C, ALPHA_C, 1);

    // variable stage 1: v1 = (1-a)*([aggPv|aggNv|v_h]@Wvcat + bV) + a*v_h0 (no relu)
    k_wmma_gemm<3><<<gGemmV, 256, 0, stream>>>(
        (const __bf16*)aggPv_b, (const __bf16*)aggNv_b, (const __bf16*)vh_b,
        (const __bf16*)(WvCat + (size_t)i * 3 * HH), biasV + (size_t)i * HD,
        v_h0, nullptr, v_tmp, v1_b, tilesV, 1.0f - ALPHA_C, ALPHA_C, 0);

    // variable stage 2: v_h = relu((1-b)*v1 + b*(v1@Wv + bv) + v_old)
    float beta = logf(THETA_C / (float)(i + 1) + 1.0f);
    k_wmma_gemm<1><<<gGemmV, 256, 0, stream>>>(
        (const __bf16*)v1_b, (const __bf16*)v1_b, (const __bf16*)v1_b,
        (const __bf16*)(Wvm + (size_t)i * HH), bv + (size_t)i * HD,
        v_tmp, v_h, v_h, vh_b, tilesV, beta, 1.0f - beta, 1);
  }

  // ---- final projection ----
  k_final<<<(NVN + TB - 1) / TB, TB, 0, stream>>>(v_h, W_fin, b_fin, (float*)d_out);
}
